// GNN_29343216566666
// MI455X (gfx1250) — matrix operand order repaired and checked by reading, not compiled, not run on hardware
//
#include <hip/hip_runtime.h>

#define N_NODES 50000
#define N_EDGES 800000
#define EMB     128
#define HID     256
#define OUT_DIM 65
#define OUT_PAD 80
#define LAYERS  5
#define N_ROOTS 2048
#define MR      32    // node rows per MLP block (2 M-tiles of 16)

typedef __attribute__((ext_vector_type(16))) __bf16 v16bf;
typedef __attribute__((ext_vector_type(8)))  __bf16 v8bf;
typedef __attribute__((ext_vector_type(4)))  __bf16 v4bf;
typedef __attribute__((ext_vector_type(8)))  float  v8f;

// ---- A fragment: 16x32 bf16 tile from row-major LDS (stride in elements) ----
// lanes 0-15: M=lane, elems 0-7 -> K=0..7, 8-15 -> K=16..23
// lanes 16-31: M=lane-16, elems 0-7 -> K=8..15, 8-15 -> K=24..31
// Both 8-element chunks are 16B-aligned contiguous -> ds_load_b128 x2.
__device__ inline v16bf load_a_frag(const __bf16* base, int stride, int lane) {
  int m = lane & 15, half = lane >> 4;
  const __bf16* row = base + m * stride;
  v8bf lo = *(const v8bf*)(row + half * 8);
  v8bf hi = *(const v8bf*)(row + 16 + half * 8);
  return __builtin_shufflevector(lo, hi, 0, 1, 2, 3, 4, 5, 6, 7,
                                 8, 9, 10, 11, 12, 13, 14, 15);
}

// ---- h0 = keys_emb[x[:,0]] + vals_emb[x[:,1]] ----
__global__ __launch_bounds__(256) void embed_kernel(
    const int* __restrict__ x, const float* __restrict__ keys,
    const float* __restrict__ vals, float* __restrict__ h0) {
  int idx = blockIdx.x * 256 + threadIdx.x;
  if (idx >= N_NODES * EMB) return;
  int i = idx >> 7, d = idx & 127;
  h0[idx] = keys[x[2 * i] * EMB + d] + vals[x[2 * i + 1] * EMB + d];
}

__global__ __launch_bounds__(256) void zero_kernel(float* __restrict__ p, int n) {
  int i = blockIdx.x * 256 + threadIdx.x;
  if (i < n) p[i] = 0.0f;
}

// Pack row-major [layers][K][N_src] fp32 weights into WMMA B-fragment order:
// dst[(((layer*NT + nt)*KT + kt)*32 + lane)*16 + i], zero-padding N to N_pad.
// Each lane's 16 bf16 fragment becomes one contiguous 32B chunk.
__global__ __launch_bounds__(256) void pack_b_kernel(
    const float* __restrict__ src, __bf16* __restrict__ dst,
    int K, int N_src, int N_pad, int layers) {
  int NT = N_pad / 16, KT = K / 32;
  int total = layers * NT * KT * 512;
  int idx = blockIdx.x * 256 + threadIdx.x;
  if (idx >= total) return;
  int i    = idx & 15;
  int lane = (idx >> 4) & 31;
  int rest = idx >> 9;
  int kt = rest % KT; rest /= KT;
  int nt = rest % NT; rest /= NT;
  int layer = rest;
  int k = kt * 32 + (lane >> 4) * 8 + (i & 7) + (i >> 3) * 16;
  int n = nt * 16 + (lane & 15);
  float v = (n < N_src) ? src[((size_t)layer * K + k) * N_src + n] : 0.0f;
  dst[idx] = (__bf16)v;
}

// ---- edge scatter-sum: one wave per edge, 32 lanes x float4 = 128 dims ----
__global__ __launch_bounds__(256) void scatter_kernel(
    const float* __restrict__ h, float* __restrict__ agg,
    const int* __restrict__ edge_index) {
  int lane = threadIdx.x & 31;
  int e = blockIdx.x * 8 + (threadIdx.x >> 5);
  if (e >= N_EDGES) return;
  int src = edge_index[e];
  int dst = edge_index[N_EDGES + e];
  const float4 v = *(const float4*)(h + src * EMB + lane * 4);
  float* a = agg + dst * EMB + lane * 4;
  atomicAdd(a + 0, v.x);
  atomicAdd(a + 1, v.y);
  atomicAdd(a + 2, v.z);
  atomicAdd(a + 3, v.w);
}

// ---- fused GIN layer: z=(1+eps)h+agg; relu(zW1+b1)W2+b2; +h; LayerNorm ----
// 32 nodes/block (2 M-tiles), 256 threads = 8 waves. Each B fragment is
// loaded once (2x global_load_b128) and reused for both M-tiles.
__global__ __launch_bounds__(256) void mlp_kernel(
    const float* __restrict__ hin, const float* __restrict__ agg,
    float* __restrict__ hout, const float* __restrict__ eps,
    const __bf16* __restrict__ W1p, const float* __restrict__ b1,
    const __bf16* __restrict__ W2p, const float* __restrict__ b2,
    const float* __restrict__ ln_g, const float* __restrict__ ln_b,
    int layer) {
  __shared__ __bf16 sZ[MR][EMB];      // GEMM1 A operand
  __shared__ float  sH[MR][EMB];      // residual
  __shared__ __bf16 sHid[MR][HID];    // GEMM2 A operand
  __shared__ float  sOut[MR][EMB];    // pre-LN result
  __shared__ float  sMean[MR], sRstd[MR];

  int tid = threadIdx.x;
  int n0  = blockIdx.x * MR;
  float e1 = 1.0f + eps[layer];

  // stage z (bf16) and h (fp32): float4 loads, packed 4xbf16 stores
  for (int idx = tid; idx < MR * (EMB / 4); idx += 256) {
    int m  = idx >> 5;            // EMB/4 == 32 chunks per row
    int d4 = (idx & 31) * 4;
    int node = n0 + m;
    int nc = node < N_NODES ? node : N_NODES - 1;   // clamp for tail block
    const float4 hv = *(const float4*)(hin + (size_t)nc * EMB + d4);
    const float4 av = *(const float4*)(agg + (size_t)nc * EMB + d4);
    *(float4*)&sH[m][d4] = hv;
    v4bf z;
    z[0] = (__bf16)(e1 * hv.x + av.x);
    z[1] = (__bf16)(e1 * hv.y + av.y);
    z[2] = (__bf16)(e1 * hv.z + av.z);
    z[3] = (__bf16)(e1 * hv.w + av.w);
    *(v4bf*)&sZ[m][d4] = z;
  }
  __syncthreads();

  int lane = tid & 31;
  int wave = tid >> 5;
  int ncol = lane & 15;
  int rowbase = (lane >> 4) * 8;   // C/D: VGPR j -> row rowbase+j

  const float* b1l = b1 + layer * HID;

  // ---- GEMM1: [32x128] @ [128x256] + b1, ReLU -> sHid ----
#pragma unroll
  for (int t = 0; t < 2; ++t) {
    int nt = wave * 2 + t;
    int c0 = nt * 16;
    float bias = b1l[c0 + ncol];
    v8f acc0, acc1;
#pragma unroll
    for (int j = 0; j < 8; ++j) { acc0[j] = bias; acc1[j] = bias; }
    const __bf16* bbase =
        W1p + ((size_t)(layer * (HID / 16) + nt) * (EMB / 32)) * 512 + lane * 16;
#pragma unroll
    for (int ks = 0; ks < EMB / 32; ++ks) {
      v16bf b  = *(const v16bf*)(bbase + ks * 512);
      v16bf a0 = load_a_frag(&sZ[0][ks * 32], EMB, lane);
      v16bf a1 = load_a_frag(&sZ[16][ks * 32], EMB, lane);
      acc0 = __builtin_amdgcn_wmma_f32_16x16x32_bf16(false, a0, false, b,
                                                     (short)0, acc0, false, false);
      acc1 = __builtin_amdgcn_wmma_f32_16x16x32_bf16(false, a1, false, b,
                                                     (short)0, acc1, false, false);
    }
#pragma unroll
    for (int j = 0; j < 8; ++j) {
      float v0 = acc0[j] > 0.0f ? acc0[j] : 0.0f;
      float v1 = acc1[j] > 0.0f ? acc1[j] : 0.0f;
      sHid[rowbase + j][c0 + ncol]      = (__bf16)v0;
      sHid[16 + rowbase + j][c0 + ncol] = (__bf16)v1;
    }
  }
  __syncthreads();

  // ---- GEMM2: [32x256] @ [256x128] + b2, +residual -> sOut ----
  const float* b2l = b2 + layer * EMB;
  {
    int c0 = wave * 16;
    float bias = b2l[c0 + ncol];
    v8f acc0, acc1;
#pragma unroll
    for (int j = 0; j < 8; ++j) { acc0[j] = bias; acc1[j] = bias; }
    const __bf16* bbase =
        W2p + ((size_t)(layer * (EMB / 16) + wave) * (HID / 32)) * 512 + lane * 16;
#pragma unroll
    for (int ks = 0; ks < HID / 32; ++ks) {
      v16bf b  = *(const v16bf*)(bbase + ks * 512);
      v16bf a0 = load_a_frag(&sHid[0][ks * 32], HID, lane);
      v16bf a1 = load_a_frag(&sHid[16][ks * 32], HID, lane);
      acc0 = __builtin_amdgcn_wmma_f32_16x16x32_bf16(false, a0, false, b,
                                                     (short)0, acc0, false, false);
      acc1 = __builtin_amdgcn_wmma_f32_16x16x32_bf16(false, a1, false, b,
                                                     (short)0, acc1, false, false);
    }
#pragma unroll
    for (int j = 0; j < 8; ++j) {
      int m0 = rowbase + j, m1 = 16 + rowbase + j;
      sOut[m0][c0 + ncol] = acc0[j] + sH[m0][c0 + ncol];
      sOut[m1][c0 + ncol] = acc1[j] + sH[m1][c0 + ncol];
    }
  }
  __syncthreads();

  // ---- LayerNorm stats: one thread per row, float4 reduction ----
  if (tid < MR) {
    const float4* row = (const float4*)&sOut[tid][0];
    float mu = 0.0f;
#pragma unroll 4
    for (int q = 0; q < EMB / 4; ++q) {
      float4 v = row[q];
      mu += v.x + v.y + v.z + v.w;
    }
    mu *= (1.0f / EMB);
    float var = 0.0f;
#pragma unroll 4
    for (int q = 0; q < EMB / 4; ++q) {
      float4 v = row[q];
      float a = v.x - mu, b = v.y - mu, c = v.z - mu, d = v.w - mu;
      var += a * a + b * b + c * c + d * d;
    }
    var *= (1.0f / EMB);
    sMean[tid] = mu;
    sRstd[tid] = rsqrtf(var + 1e-5f);
  }
  __syncthreads();

  const float* gl = ln_g + layer * EMB;
  const float* bl = ln_b + layer * EMB;
  for (int idx = tid; idx < MR * (EMB / 4); idx += 256) {
    int m  = idx >> 5;
    int d4 = (idx & 31) * 4;
    int node = n0 + m;
    if (node >= N_NODES) continue;
    float mu = sMean[m], rs = sRstd[m];
    float4 o;
    o.x = (sOut[m][d4 + 0] - mu) * rs * gl[d4 + 0] + bl[d4 + 0];
    o.y = (sOut[m][d4 + 1] - mu) * rs * gl[d4 + 1] + bl[d4 + 1];
    o.z = (sOut[m][d4 + 2] - mu) * rs * gl[d4 + 2] + bl[d4 + 2];
    o.w = (sOut[m][d4 + 3] - mu) * rs * gl[d4 + 3] + bl[d4 + 3];
    *(float4*)(hout + (size_t)node * EMB + d4) = o;
  }
}

// ---- readout: h[roots] @ W_out  (16 roots/block, 5 waves, packed B) ----
__global__ __launch_bounds__(160) void readout_kernel(
    const float* __restrict__ h, const int* __restrict__ roots,
    const __bf16* __restrict__ Woutp, float* __restrict__ out) {
  __shared__ __bf16 sA[16][EMB];
  int tid = threadIdx.x;
  int r0  = blockIdx.x * 16;

  for (int idx = tid; idx < 16 * (EMB / 4); idx += 160) {
    int m  = idx >> 5;
    int d4 = (idx & 31) * 4;
    const float4 hv = *(const float4*)(h + (size_t)roots[r0 + m] * EMB + d4);
    v4bf a;
    a[0] = (__bf16)hv.x; a[1] = (__bf16)hv.y;
    a[2] = (__bf16)hv.z; a[3] = (__bf16)hv.w;
    *(v4bf*)&sA[m][d4] = a;
  }
  __syncthreads();

  int lane = tid & 31;
  int wave = tid >> 5;
  int ncol = lane & 15;
  int rowbase = (lane >> 4) * 8;
  int c0 = wave * 16;

  v8f acc;
#pragma unroll
  for (int j = 0; j < 8; ++j) acc[j] = 0.0f;
  const __bf16* bbase = Woutp + ((size_t)wave * (EMB / 32)) * 512 + lane * 16;
#pragma unroll
  for (int ks = 0; ks < EMB / 32; ++ks) {
    v16bf b = *(const v16bf*)(bbase + ks * 512);
    v16bf a = load_a_frag(&sA[0][ks * 32], EMB, lane);
    acc = __builtin_amdgcn_wmma_f32_16x16x32_bf16(false, a, false, b,
                                                  (short)0, acc, false, false);
  }
  int col = c0 + ncol;
  if (col < OUT_DIM) {
#pragma unroll
    for (int j = 0; j < 8; ++j)
      out[(r0 + rowbase + j) * OUT_DIM + col] = acc[j];
  }
}

extern "C" void kernel_launch(void* const* d_in, const int* in_sizes, int n_in,
                              void* d_out, int out_size, void* d_ws, size_t ws_size,
                              hipStream_t stream) {
  (void)in_sizes; (void)n_in; (void)out_size; (void)ws_size;
  const int*   x          = (const int*)d_in[0];
  const int*   edge_index = (const int*)d_in[1];
  const int*   roots      = (const int*)d_in[2];
  const float* keys       = (const float*)d_in[3];
  const float* vals       = (const float*)d_in[4];
  const float* eps        = (const float*)d_in[5];
  const float* W1         = (const float*)d_in[6];
  const float* b1         = (const float*)d_in[7];
  const float* W2         = (const float*)d_in[8];
  const float* b2         = (const float*)d_in[9];
  const float* ln_g       = (const float*)d_in[10];
  const float* ln_b       = (const float*)d_in[11];
  const float* W_out      = (const float*)d_in[12];
  float* out = (float*)d_out;

  char* ws = (char*)d_ws;
  auto alignUp = [](size_t o) { return (o + 255) & ~(size_t)255; };
  size_t off = 0;
  float*  hA    = (float*)(ws + off);  off = alignUp(off + (size_t)N_NODES * EMB * 4);
  float*  hB    = (float*)(ws + off);  off = alignUp(off + (size_t)N_NODES * EMB * 4);
  float*  agg   = (float*)(ws + off);  off = alignUp(off + (size_t)N_NODES * EMB * 4);
  const int W1P_N = LAYERS * (HID / 16) * (EMB / 32) * 512;  // 163840
  const int W2P_N = LAYERS * (EMB / 16) * (HID / 32) * 512;  // 163840
  const int WOP_N = (OUT_PAD / 16) * (EMB / 32) * 512;       // 10240
  __bf16* w1p   = (__bf16*)(ws + off); off = alignUp(off + (size_t)W1P_N * 2);
  __bf16* w2p   = (__bf16*)(ws + off); off = alignUp(off + (size_t)W2P_N * 2);
  __bf16* woutp = (__bf16*)(ws + off); off = alignUp(off + (size_t)WOP_N * 2);

  const int NE = N_NODES * EMB;
  embed_kernel<<<(NE + 255) / 256, 256, 0, stream>>>(x, keys, vals, hA);
  pack_b_kernel<<<(W1P_N + 255) / 256, 256, 0, stream>>>(W1, w1p, EMB, HID, HID, LAYERS);
  pack_b_kernel<<<(W2P_N + 255) / 256, 256, 0, stream>>>(W2, w2p, HID, EMB, EMB, LAYERS);
  pack_b_kernel<<<(WOP_N + 255) / 256, 256, 0, stream>>>(W_out, woutp, EMB, OUT_DIM, OUT_PAD, 1);

  float* hcur = hA;
  float* hnext = hB;
  for (int l = 0; l < LAYERS; ++l) {
    zero_kernel<<<(NE + 255) / 256, 256, 0, stream>>>(agg, NE);
    scatter_kernel<<<(N_EDGES + 7) / 8, 256, 0, stream>>>(hcur, agg, edge_index);
    mlp_kernel<<<(N_NODES + MR - 1) / MR, 256, 0, stream>>>(
        hcur, agg, hnext, eps, w1p, b1, w2p, b2, ln_g, ln_b, l);
    float* t = hcur; hcur = hnext; hnext = t;
  }
  readout_kernel<<<N_ROOTS / 16, 160, 0, stream>>>(hcur, roots, woutp, out);
}
